// RoPEMHA_1520418423171
// MI455X (gfx1250) — matrix operand order repaired and checked by reading, not compiled, not run on hardware
//
#include <hip/hip_runtime.h>
#include <hip/hip_bf16.h>
#include <stdint.h>

// B=2, S=2048, D=1024, 16 heads, hd=64. fp32 in/out, bf16 WMMA internally.
#define S_LEN  2048
#define DMODEL 1024
#define NHEADS 16
#define HDIM   64
#define BATCH  2
#define LDSPAD 40   // row stride (shorts): 80B -> 16B-aligned rows for b128

typedef __attribute__((ext_vector_type(16))) __bf16 v16bf;
typedef __attribute__((ext_vector_type(8)))  float  v8f;
typedef __attribute__((ext_vector_type(2)))  unsigned int v2u;
typedef __attribute__((ext_vector_type(4)))  unsigned int v4u;

typedef union { v16bf v; v4u q[2]; } bfvec;

// Packed fp32 -> 2x bf16 (RNE) in one VALU op.
__device__ __forceinline__ unsigned int pack2(float lo, float hi) {
  unsigned int r;
  asm("v_cvt_pk_bf16_f32 %0, %1, %2" : "=v"(r) : "v"(lo), "v"(hi));
  return r;
}
__device__ __forceinline__ unsigned short f2bf(float f) {
  return (unsigned short)pack2(f, f);
}

// A-matrix 16x32 bf16 fragment (ISA 7.12.2): lane&15 = M row; per lane the
// data is two contiguous 16B chunks: K in {0..7,16..23} (+8 for lanes>=16).
__device__ __forceinline__ v16bf frag_ld_a(const unsigned short* p, int stride) {
  int lane = threadIdx.x & 31;
  const unsigned short* rowp = p + (lane & 15) * stride + (lane >> 4) * 8;
  bfvec f;
  f.q[0] = *(const v4u*)(rowp);        // K = kh*8 + 0..7
  f.q[1] = *(const v4u*)(rowp + 16);   // K = 16 + kh*8 + 0..7
  return f.v;
}

// B-matrix 32x16 bf16 fragment: lane&15 = N col; per lane two contiguous
// 16B chunks: K in {0..7,16..23} (+8 for lanes>=16). Gathered from K-contiguous tile BsT[n][k].
__device__ __forceinline__ v16bf frag_ld_b(const unsigned short* p, int stride) {
  int lane = threadIdx.x & 31;
  const unsigned short* rowp = p + (lane & 15) * stride + (lane >> 4) * 8;
  bfvec f;
  f.q[0] = *(const v4u*)(rowp);
  f.q[1] = *(const v4u*)(rowp + 16);
  return f.v;
}

__device__ __forceinline__ v8f wmma_bf16(v16bf a, v16bf b, v8f c) {
  return __builtin_amdgcn_wmma_f32_16x16x32_bf16(false, a, false, b,
                                                 (short)0, c, false, false);
}

__device__ __forceinline__ float rmax16(float v) {
#pragma unroll
  for (int m = 1; m < 16; m <<= 1) v = fmaxf(v, __shfl_xor(v, m, 16));
  return v;
}
__device__ __forceinline__ float rsum16(float v) {
#pragma unroll
  for (int m = 1; m < 16; m <<= 1) v += __shfl_xor(v, m, 16);
  return v;
}

// ---------------------------------------------------------------------------
// Kernel 1: fused QKV projection (bf16 WMMA, fp32 accum) + interleaved RoPE.
// Grid: (4096/32, 1024/64). Block: 128 (4 waves, 2x2 over a 32x64 tile).
// Writes q/k bf16 as [b*16+h][s][64]; v TRANSPOSED as [b*16+h][d][s].
// ---------------------------------------------------------------------------
__global__ __launch_bounds__(128) void qkv_rope_kernel(
    const float* __restrict__ x, const float* __restrict__ Wq,
    const float* __restrict__ Wk, const float* __restrict__ Wv,
    unsigned short* __restrict__ qws, unsigned short* __restrict__ kws,
    unsigned short* __restrict__ vws) {
  __shared__ alignas(16) unsigned short As[32][LDSPAD];     // x tile   [m][k]
  __shared__ alignas(16) unsigned short Bs[3][64][LDSPAD];  // W^T tile [n][k]
  const float* Wlist[3] = {Wq, Wk, Wv};
  const int tid = threadIdx.x;
  const int m0 = blockIdx.x * 32, n0 = blockIdx.y * 64;
  const int wid = tid >> 5, lane = tid & 31;
  const int mr = wid >> 1, nc = wid & 1;
  const int lc = lane & 15, kh = lane >> 4;
  const int sn = tid & 63, skg0 = tid >> 6;  // W staging: n-row / k-group split
  v8f acc[3][2] = {};

  for (int kk = 0; kk < DMODEL; kk += 32) {
    // x tile: 256 float4 tasks, 2 per thread -> b128 load + b64 LDS store
#pragma unroll
    for (int it = 0; it < 2; ++it) {
      int idx = tid + it * 128;
      int r = idx >> 3, c = (idx & 7) * 4;
      const float4 xv = *(const float4*)(&x[(size_t)(m0 + r) * DMODEL + kk + c]);
      v2u pk; pk.x = pack2(xv.x, xv.y); pk.y = pack2(xv.z, xv.w);
      *(v2u*)(&As[r][c]) = pk;
    }
    // W tiles transposed: thread owns n-row, reads 4 strided fp32, b64 store
#pragma unroll
    for (int w = 0; w < 3; ++w) {
      const float* W = Wlist[w] + (size_t)kk * DMODEL + n0 + sn;
#pragma unroll
      for (int kg = skg0; kg < 8; kg += 2) {
        float a0 = W[(kg * 4 + 0) * DMODEL];
        float a1 = W[(kg * 4 + 1) * DMODEL];
        float a2 = W[(kg * 4 + 2) * DMODEL];
        float a3 = W[(kg * 4 + 3) * DMODEL];
        v2u pk; pk.x = pack2(a0, a1); pk.y = pack2(a2, a3);
        *(v2u*)(&Bs[w][sn][kg * 4]) = pk;
      }
    }
    if (kk + 32 < DMODEL)
      __builtin_prefetch(&x[(size_t)(m0 + (tid >> 2)) * DMODEL + kk + 32], 0, 1);
    __syncthreads();

    // Issue ALL fragment loads first so DS returns overlap WMMA execution.
    v16bf a = frag_ld_a(&As[mr * 16][0], LDSPAD);
    v16bf bfr[6];
#pragma unroll
    for (int w = 0; w < 3; ++w)
#pragma unroll
      for (int t = 0; t < 2; ++t)
        bfr[w * 2 + t] = frag_ld_b(&Bs[w][nc * 32 + t * 16][0], LDSPAD);
#pragma unroll
    for (int w = 0; w < 3; ++w)
#pragma unroll
      for (int t = 0; t < 2; ++t)
        acc[w][t] = wmma_bf16(a, bfr[w * 2 + t], acc[w][t]);
    __syncthreads();
  }

  // Epilogue: RoPE on q,k (pairs are adjacent columns -> adjacent lanes).
#pragma unroll
  for (int w = 0; w < 3; ++w) {
#pragma unroll
    for (int t = 0; t < 2; ++t) {
      v8f c = acc[w][t];
      int n_glob = n0 + nc * 32 + t * 16 + lc;
      int h = n_glob >> 6, d = n_glob & 63;
      if (w < 2) {
        float freq = __powf(10000.0f, -(float)(d >> 1) * (1.0f / 32.0f));
#pragma unroll
        for (int j = 0; j < 8; ++j) {
          int mg = m0 + mr * 16 + j + 8 * kh;
          float s_, c_;
          __sincosf((float)(mg & (S_LEN - 1)) * freq, &s_, &c_);
          float self = c[j];
          float pv = __shfl_xor(self, 1, 32);      // paired even/odd element
          c[j] = (d & 1) ? (pv * s_ + self * c_)   // odd:  e*sin + o*cos
                         : (self * c_ - pv * s_);  // even: e*cos - o*sin
        }
#pragma unroll
        for (int j = 0; j < 8; ++j) {
          int mg = m0 + mr * 16 + j + 8 * kh;
          int b = mg >> 11, s = mg & (S_LEN - 1);
          unsigned short* dst = (w == 0) ? qws : kws;
          dst[(((size_t)(b * NHEADS + h) * S_LEN + s) << 6) + d] = f2bf(c[j]);
        }
      } else {
        // V transposed [bh][d][s]: 8 consecutive s per thread -> one b128
        int mg0 = m0 + mr * 16 + 8 * kh;
        int b = mg0 >> 11, s0 = mg0 & (S_LEN - 1);
        int bh = b * NHEADS + h;
        v4u pk;
        pk.x = pack2(c[0], c[1]); pk.y = pack2(c[2], c[3]);
        pk.z = pack2(c[4], c[5]); pk.w = pack2(c[6], c[7]);
        *(v4u*)(&vws[((size_t)bh * HDIM + d) * S_LEN + s0]) = pk;
      }
    }
  }
}

// ---------------------------------------------------------------------------
// Kernel 2: causal flash attention, one wave per 16 query rows.
// Grid: (2048/64, 32). Block: 128. Online softmax over 32-key blocks.
// All Q/K/V fragments are 2x b128 global loads (L2-resident bf16).
// ---------------------------------------------------------------------------
__global__ __launch_bounds__(128) void flash_kernel(
    const unsigned short* __restrict__ qws, const unsigned short* __restrict__ kws,
    const unsigned short* __restrict__ vws, unsigned short* __restrict__ ctxws) {
  __shared__ alignas(16) unsigned short Ps[4][16][32];  // per-wave P staging
  const int tid = threadIdx.x, wid = tid >> 5, lane = tid & 31;
  const int lc = lane & 15, kh = lane >> 4;
  const int bh = blockIdx.y, b = bh >> 4, h = bh & (NHEADS - 1);
  const unsigned short* qp = qws + (size_t)bh * S_LEN * HDIM;
  const unsigned short* kp = kws + (size_t)bh * S_LEN * HDIM;
  const unsigned short* vp = vws + (size_t)bh * HDIM * S_LEN;  // [d][s]
  const int q0 = blockIdx.x * 64 + wid * 16;

  v16bf aq0 = frag_ld_a(qp + q0 * HDIM, HDIM);       // K-dim d = 0..31
  v16bf aq1 = frag_ld_a(qp + q0 * HDIM + 32, HDIM);  // K-dim d = 32..63
  v8f ctx[4] = {};
  float mrow[8], lrow[8];
#pragma unroll
  for (int j = 0; j < 8; ++j) { mrow[j] = -3.0e38f; lrow[j] = 0.0f; }

  for (int kb = 0; kb <= q0 + 15; kb += 32) {
    // Issue all 4 K B-frag loads, then the score WMMAs.
    v16bf bk00 = frag_ld_b(kp + kb * HDIM, HDIM);
    v16bf bk01 = frag_ld_b(kp + kb * HDIM + 32, HDIM);
    v16bf bk10 = frag_ld_b(kp + (kb + 16) * HDIM, HDIM);
    v16bf bk11 = frag_ld_b(kp + (kb + 16) * HDIM + 32, HDIM);
    v8f z = {};
    v8f s0 = wmma_bf16(aq1, bk01, wmma_bf16(aq0, bk00, z));
    v8f s1 = wmma_bf16(aq1, bk11, wmma_bf16(aq0, bk10, z));

    float resc[8];
#pragma unroll
    for (int j = 0; j < 8; ++j) {
      int row = q0 + j + 8 * kh;
      float v0 = (kb + lc      <= row) ? s0[j] * 0.125f : -3.0e38f;
      float v1 = (kb + 16 + lc <= row) ? s1[j] * 0.125f : -3.0e38f;
      float rmax = rmax16(fmaxf(v0, v1));           // one tree for 32 cols
      float mnew = fmaxf(mrow[j], rmax);
      resc[j] = __expf(mrow[j] - mnew);
      float p0 = __expf(v0 - mnew);
      float p1 = __expf(v1 - mnew);
      lrow[j] = lrow[j] * resc[j] + rsum16(p0 + p1);
      mrow[j] = mnew;
      Ps[wid][j + 8 * kh][lc]      = f2bf(p0);
      Ps[wid][j + 8 * kh][16 + lc] = f2bf(p1);
    }
#pragma unroll
    for (int dt = 0; dt < 4; ++dt)
#pragma unroll
      for (int j = 0; j < 8; ++j) ctx[dt][j] *= resc[j];

    // Issue all 4 V^T B-frag loads (global) before the Ps gather + WMMAs.
    v16bf bv[4];
#pragma unroll
    for (int dt = 0; dt < 4; ++dt)
      bv[dt] = frag_ld_b(vp + (size_t)(dt * 16) * S_LEN + kb, S_LEN);

    asm volatile("s_wait_dscnt 0" ::: "memory");  // Ps visible within wave
    v16bf ap = frag_ld_a(&Ps[wid][0][0], 32);
#pragma unroll
    for (int dt = 0; dt < 4; ++dt)
      ctx[dt] = wmma_bf16(ap, bv[dt], ctx[dt]);
  }

#pragma unroll
  for (int j = 0; j < 8; ++j) {
    float inv = 1.0f / lrow[j];
    int row = q0 + j + 8 * kh;
#pragma unroll
    for (int dt = 0; dt < 4; ++dt) {
      int col = h * HDIM + dt * 16 + lc;
      ctxws[((size_t)b * S_LEN + row) * DMODEL + col] = f2bf(ctx[dt][j] * inv);
    }
  }
}

// ---------------------------------------------------------------------------
// Kernel 3: output projection ctx @ Wo + bo, fp32 output.
// ---------------------------------------------------------------------------
__global__ __launch_bounds__(128) void oproj_kernel(
    const unsigned short* __restrict__ ctxws, const float* __restrict__ Wo,
    const float* __restrict__ bo, float* __restrict__ out) {
  __shared__ alignas(16) unsigned short Bs[64][LDSPAD];  // Wo^T tile [n][k]
  const int tid = threadIdx.x, wid = tid >> 5, lane = tid & 31;
  const int mr = wid >> 1, nc = wid & 1;
  const int lc = lane & 15, kh = lane >> 4;
  const int m0 = blockIdx.x * 32, n0 = blockIdx.y * 64;
  const int sn = tid & 63, skg0 = tid >> 6;
  v8f acc[2] = {};

  for (int kk = 0; kk < DMODEL; kk += 32) {
    const float* W = Wo + (size_t)kk * DMODEL + n0 + sn;
#pragma unroll
    for (int kg = skg0; kg < 8; kg += 2) {
      float a0 = W[(kg * 4 + 0) * DMODEL];
      float a1 = W[(kg * 4 + 1) * DMODEL];
      float a2 = W[(kg * 4 + 2) * DMODEL];
      float a3 = W[(kg * 4 + 3) * DMODEL];
      v2u pk; pk.x = pack2(a0, a1); pk.y = pack2(a2, a3);
      *(v2u*)(&Bs[sn][kg * 4]) = pk;
    }
    __syncthreads();
    // A-frag directly from bf16 ctx in global (L2-resident): 2x b128
    v16bf a = frag_ld_a(ctxws + (size_t)(m0 + mr * 16) * DMODEL + kk, DMODEL);
    v16bf b0 = frag_ld_b(&Bs[nc * 32][0], LDSPAD);
    v16bf b1 = frag_ld_b(&Bs[nc * 32 + 16][0], LDSPAD);
    acc[0] = wmma_bf16(a, b0, acc[0]);
    acc[1] = wmma_bf16(a, b1, acc[1]);
    __syncthreads();
  }

#pragma unroll
  for (int t = 0; t < 2; ++t) {
    int n_glob = n0 + nc * 32 + t * 16 + lc;
    float bias = bo[n_glob];
#pragma unroll
    for (int j = 0; j < 8; ++j) {
      int mg = m0 + mr * 16 + j + 8 * kh;
      out[(size_t)mg * DMODEL + n_glob] = acc[t][j] + bias;
    }
  }
}

// ---------------------------------------------------------------------------
extern "C" void kernel_launch(void* const* d_in, const int* in_sizes, int n_in,
                              void* d_out, int out_size, void* d_ws, size_t ws_size,
                              hipStream_t stream) {
  const float* x  = (const float*)d_in[0];
  const float* Wq = (const float*)d_in[1];
  const float* Wk = (const float*)d_in[2];
  const float* Wv = (const float*)d_in[3];
  const float* Wo = (const float*)d_in[4];
  const float* bo = (const float*)d_in[5];

  const size_t perTensor = (size_t)BATCH * NHEADS * S_LEN * HDIM;  // 4.19M elems
  unsigned short* qws = (unsigned short*)d_ws;
  unsigned short* kws = qws + perTensor;
  unsigned short* vws = kws + perTensor;   // transposed [bh][d][s]
  unsigned short* cws = vws + perTensor;   // ctx as [b][s][1024] bf16

  dim3 blk(128);
  qkv_rope_kernel<<<dim3(BATCH * S_LEN / 32, DMODEL / 64), blk, 0, stream>>>(
      x, Wq, Wk, Wv, qws, kws, vws);
  flash_kernel<<<dim3(S_LEN / 64, BATCH * NHEADS), blk, 0, stream>>>(
      qws, kws, vws, cws);
  oproj_kernel<<<dim3(BATCH * S_LEN / 32, DMODEL / 64), blk, 0, stream>>>(
      cws, Wo, bo, (float*)d_out);
}
